// MambaEnc_84997402787919
// MI455X (gfx1250) — matrix operand order repaired and checked by reading, not compiled, not run on hardware
//
#include <hip/hip_runtime.h>
#include <hip/hip_bf16.h>

// ---------------------------------------------------------------------------
// Mamba encoder forward for MI455X (gfx1250, wave32, WMMA + TDM).
// All GEMM operands pre-converted to f16; v_wmma_f32_16x16x32_f16 with f32
// accumulate; LDS fed by tensor_load_to_lds (double buffered, TENSORcnt)
// when the TDM builtin is available, manual f16 loads otherwise.
// ---------------------------------------------------------------------------

typedef __attribute__((ext_vector_type(16))) _Float16 v16h;
typedef __attribute__((ext_vector_type(8)))  _Float16 v8h;
typedef __attribute__((ext_vector_type(8)))  float    v8f;
typedef __attribute__((ext_vector_type(4)))  unsigned u32x4;
typedef __attribute__((ext_vector_type(8)))  int      i32x8;
typedef __attribute__((ext_vector_type(4)))  int      i32x4;

#define BATCH   4
#define SEQ     1024
#define IN_DIM  35
#define DMODEL  512
#define DINNER  1024
#define NSTATE  16
#define DTRANK  32
#define DCONV   4
#define NROWS   (BATCH * SEQ)   // 4096

#if defined(__has_builtin)
#  if __has_builtin(__builtin_amdgcn_tensor_load_to_lds) && \
      __has_builtin(__builtin_amdgcn_s_wait_tensorcnt)
#    define HAVE_TDM 1
#  endif
#endif
#ifndef HAVE_TDM
#  define HAVE_TDM 0
#endif

// ---------------------------------------------------------------------------
// f32 -> f16 bulk convert (weights, once per launch; everything stays in L2)
// ---------------------------------------------------------------------------
__global__ __launch_bounds__(256)
void f32_to_f16_kernel(const float* __restrict__ in, _Float16* __restrict__ out, int n) {
    const int i = blockIdx.x * 256 + threadIdx.x;
    if (i < n) out[i] = (_Float16)in[i];
}

// ---------------------------------------------------------------------------
// Embedding: hs[row, o] = sum_k x[row, k] * W[o, k] + b[o]   (K = 35)
// ---------------------------------------------------------------------------
__global__ __launch_bounds__(512)
void embed_kernel(const float* __restrict__ x, const float* __restrict__ w,
                  const float* __restrict__ b, float* __restrict__ hs) {
    const int row = blockIdx.x;
    __shared__ float xr[IN_DIM + 1];
    if (threadIdx.x < IN_DIM) xr[threadIdx.x] = x[row * IN_DIM + threadIdx.x];
    __syncthreads();
    const int o = threadIdx.x;
    float acc = b[o];
    #pragma unroll
    for (int k = 0; k < IN_DIM; ++k) acc += xr[k] * w[o * IN_DIM + k];
    hs[row * DMODEL + o] = acc;
}

// ---------------------------------------------------------------------------
// residual (+=) then RMSNorm -> hn written directly as f16 (GEMM A operand).
// ---------------------------------------------------------------------------
__global__ __launch_bounds__(256)
void add_rmsnorm_kernel(const float* __restrict__ hs, float* __restrict__ res,
                        _Float16* __restrict__ hn, const float* __restrict__ w,
                        int first) {
    const int row = blockIdx.x;
    const int t   = threadIdx.x;
    const float* hrow = hs + (size_t)row * DMODEL;
    float* rrow = res + (size_t)row * DMODEL;
    float v0 = hrow[t];
    float v1 = hrow[t + 256];
    if (!first) { v0 += rrow[t]; v1 += rrow[t + 256]; }
    rrow[t] = v0; rrow[t + 256] = v1;

    __shared__ float red[256];
    red[t] = v0 * v0 + v1 * v1;
    __syncthreads();
    #pragma unroll
    for (int s = 128; s > 0; s >>= 1) {
        if (t < s) red[t] += red[t + s];
        __syncthreads();
    }
    const float scale = rsqrtf(red[0] * (1.0f / (float)DMODEL) + 1e-5f);
    hn[(size_t)row * DMODEL + t]       = (_Float16)(v0 * scale * w[t]);
    hn[(size_t)row * DMODEL + t + 256] = (_Float16)(v1 * scale * w[t + 256]);
}

#if HAVE_TDM
// ---------------------------------------------------------------------------
// Issue one 2D TDM tile load: global (rows x 32 f16, row stride lda elems)
// -> LDS at byte offset lds_off with 48-half row stride via pad feature
// (pad_enable, 8 dwords pad per 16 dwords stored). D# per ISA 08 §8.3/8.4.
// ---------------------------------------------------------------------------
__device__ __forceinline__
void tdm_load_2d(const _Float16* gptr, unsigned lds_off, int Kdim, int rows,
                 int lda, int tile_rows) {
    const unsigned long long ga = (unsigned long long)(uintptr_t)gptr;
    u32x4 g0;
    g0[0] = 1u;                                   // count=1, user mode
    g0[1] = lds_off;                              // lds_addr
    g0[2] = (unsigned)ga;                         // global_addr[31:0]
    g0[3] = (unsigned)((ga >> 32) & 0x01FFFFFFu)  // global_addr[56:32]
          | (2u << 30);                           // type = 2 ("image")
    i32x8 g1;
    g1[0] = (int)((1u << 16)                      // data_size = 2 bytes
          | (1u << 20)                            // pad_enable
          | (3u << 22)                            // pad_interval: 16 dwords
          | (7u << 25));                          // pad_amount:   8 dwords
    g1[1] = (int)((unsigned)(Kdim & 0xFFFF) << 16);            // tensor_dim0 lo
    g1[2] = (int)(((unsigned)(Kdim >> 16) & 0xFFFFu)
          | ((unsigned)(rows & 0xFFFF) << 16));                // dim0 hi | dim1 lo
    g1[3] = (int)(((unsigned)(rows >> 16) & 0xFFFFu)
          | (32u << 16));                                      // dim1 hi | tile_dim0=32
    g1[4] = tile_rows;                                         // tile_dim1
    g1[5] = lda;                                               // tensor_dim0_stride lo
    g1[6] = 0;
    g1[7] = 0;
    i32x4 z4 = {0, 0, 0, 0};
    i32x8 z8 = {0, 0, 0, 0, 0, 0, 0, 0};
    __builtin_amdgcn_tensor_load_to_lds(g0, g1, z4, z4, z8, 0);
}
#endif

// ---------------------------------------------------------------------------
// WMMA GEMM: C[M,N] = act( A[M,K]f16 @ W[N,K]^T f16 + bias ), M = 4096.
// Block tile 128x64 (8 waves); each wave owns a 32x32 register tile =
// 2x2 v_wmma_f32_16x16x32_f16 per K-step (each fragment reused twice).
// Double-buffered LDS, filled by TDM (wave 0 issues, s_wait_tensorcnt
// pipelined) or by manual b128 f16 loads as fallback.
// ACT: 0 = none, 1 = softplus. Cf16: optional secondary f16 output.
// ---------------------------------------------------------------------------
template <int ACT>
__global__ __launch_bounds__(256)
void gemm_wmma(const _Float16* __restrict__ A, int lda,
               const _Float16* __restrict__ W, int ldw,
               float* __restrict__ C, int ldc, int K,
               const float* __restrict__ bias,
               _Float16* __restrict__ Cf16) {
    __shared__ alignas(16) _Float16 As[2][128][48];  // 32 K + pad -> 96B rows
    __shared__ alignas(16) _Float16 Bs[2][64][48];

    const int tid  = threadIdx.x;
    const int wave = tid >> 5;
    const int lane = tid & 31;
    const int m0 = blockIdx.x * 128;
    const int n0 = blockIdx.y * 64;
    const int wr = (wave >> 1) * 32;   // wave row sub-tile: 0/32/64/96
    const int wc = (wave & 1) * 32;    // wave col sub-tile: 0/32

    v8f acc00 = {}, acc01 = {}, acc10 = {}, acc11 = {};
    const int nk = K / 32;

#if HAVE_TDM
    const unsigned asOff[2] = { (unsigned)(uintptr_t)&As[0][0][0],
                                (unsigned)(uintptr_t)&As[1][0][0] };
    const unsigned bsOff[2] = { (unsigned)(uintptr_t)&Bs[0][0][0],
                                (unsigned)(uintptr_t)&Bs[1][0][0] };
    if (wave == 0) {   // TDM ignores EXEC; one issue per wave, wave 0 only
        tdm_load_2d(A + (size_t)m0 * lda, asOff[0], K, 128, lda, 128);
        tdm_load_2d(W + (size_t)n0 * ldw, bsOff[0], K, 64,  ldw, 64);
    }
#else
    // manual cooperative staging mappings (f16 -> LDS, no conversion needed)
    const int ar = tid >> 1;            // 0..127
    const int ak = (tid & 1) * 16;      // 0 / 16
    const int br = tid >> 2;            // 0..63
    const int bk = (tid & 3) * 8;       // 0..24
    {   // prologue: stage 0 -> buffer 0
        v8h a0 = *(const v8h*)(A + (size_t)(m0 + ar) * lda + ak);
        v8h a1 = *(const v8h*)(A + (size_t)(m0 + ar) * lda + ak + 8);
        v8h b0 = *(const v8h*)(W + (size_t)(n0 + br) * ldw + bk);
        *(v8h*)&As[0][ar][ak]     = a0;
        *(v8h*)&As[0][ar][ak + 8] = a1;
        *(v8h*)&Bs[0][br][bk]     = b0;
    }
#endif

    for (int i = 0; i < nk; ++i) {
        const int cur = i & 1;
        const int k0n = (i + 1) * 32;

#if HAVE_TDM
        if (i + 1 < nk) {
            if (wave == 0) {
                tdm_load_2d(A + (size_t)m0 * lda + k0n, asOff[cur ^ 1], K, 128, lda, 128);
                tdm_load_2d(W + (size_t)n0 * ldw + k0n, bsOff[cur ^ 1], K, 64,  ldw, 64);
            }
            __builtin_amdgcn_s_wait_tensorcnt(2);   // current stage's pair done
        } else {
            __builtin_amdgcn_s_wait_tensorcnt(0);
        }
        __syncthreads();
#else
        v8h na0, na1, nb0;
        if (i + 1 < nk) {
            na0 = *(const v8h*)(A + (size_t)(m0 + ar) * lda + k0n + ak);
            na1 = *(const v8h*)(A + (size_t)(m0 + ar) * lda + k0n + ak + 8);
            nb0 = *(const v8h*)(W + (size_t)(n0 + br) * ldw + k0n + bk);
            __builtin_prefetch(A + (size_t)(m0 + ar) * lda + k0n + 32, 0, 3);
        }
        __syncthreads();
#endif

        // --- fragments (ISA 7.12.2 layouts) ---
        // A: lanes 0-15 row M=lane%16, halves K[0..7],K[16..23];
        //    lanes 16-31 same row set, halves K[8..15],K[24..31].
        const int arow = wr + (lane & 15);
        const int akb  = (lane < 16) ? 0 : 8;
        v16h afr[2];
        #pragma unroll
        for (int t2 = 0; t2 < 2; ++t2) {
            v8h lo = *(const v8h*)&As[cur][arow + t2 * 16][akb];
            v8h hi = *(const v8h*)&As[cur][arow + t2 * 16][akb + 16];
            #pragma unroll
            for (int q = 0; q < 8; ++q) { afr[t2][q] = lo[q]; afr[t2][q + 8] = hi[q]; }
        }
        // B: lane holds column N=lane%16; K[0..7],K[16..23] (lanes 0-15) / K[8..15],K[24..31].
        const int bcol = wc + (lane & 15);
        const int bkb  = (lane < 16) ? 0 : 8;
        v16h bfr[2];
        #pragma unroll
        for (int t2 = 0; t2 < 2; ++t2) {
            v8h lo = *(const v8h*)&Bs[cur][bcol + t2 * 16][bkb];
            v8h hi = *(const v8h*)&Bs[cur][bcol + t2 * 16][bkb + 16];
            #pragma unroll
            for (int q = 0; q < 8; ++q) { bfr[t2][q] = lo[q]; bfr[t2][q + 8] = hi[q]; }
        }

        acc00 = __builtin_amdgcn_wmma_f32_16x16x32_f16(false, afr[0], false, bfr[0], (short)0, acc00, false, false);
        acc01 = __builtin_amdgcn_wmma_f32_16x16x32_f16(false, afr[0], false, bfr[1], (short)0, acc01, false, false);
        acc10 = __builtin_amdgcn_wmma_f32_16x16x32_f16(false, afr[1], false, bfr[0], (short)0, acc10, false, false);
        acc11 = __builtin_amdgcn_wmma_f32_16x16x32_f16(false, afr[1], false, bfr[1], (short)0, acc11, false, false);

#if !HAVE_TDM
        if (i + 1 < nk) {
            *(v8h*)&As[cur ^ 1][ar][ak]     = na0;
            *(v8h*)&As[cur ^ 1][ar][ak + 8] = na1;
            *(v8h*)&Bs[cur ^ 1][br][bk]     = nb0;
        }
#endif
        __syncthreads();
    }

    // D layout: VGPR r -> row r (lanes 0-15) / r+8 (lanes 16-31); col = lane%16
    auto store_tile = [&](const v8f& a, int ti, int tj) {
        const int col = n0 + wc + tj + (lane & 15);
        const float bv = bias ? bias[col] : 0.0f;
        #pragma unroll
        for (int r = 0; r < 8; ++r) {
            const int row = m0 + wr + ti + ((lane < 16) ? r : r + 8);
            float v = a[r] + bv;
            if (ACT == 1) v = (v > 20.0f) ? v : __logf(1.0f + __expf(v));
            C[(size_t)row * ldc + col] = v;
            if (Cf16) Cf16[(size_t)row * ldc + col] = (_Float16)v;
        }
    };
    store_tile(acc00, 0, 0);
    store_tile(acc01, 0, 16);
    store_tile(acc10, 16, 0);
    store_tile(acc11, 16, 16);
}

// ---------------------------------------------------------------------------
// Depthwise causal conv1d (kernel 4) + SiLU; writes f32 (scan u) + f16 (GEMM A).
// ---------------------------------------------------------------------------
__global__ __launch_bounds__(256)
void conv_silu_kernel(const float* __restrict__ xz, const float* __restrict__ cw,
                      const float* __restrict__ cb, float* __restrict__ xc,
                      _Float16* __restrict__ xch) {
    const int idx = blockIdx.x * blockDim.x + threadIdx.x; // B*T*DINNER
    const int d  = idx & (DINNER - 1);
    const int bt = idx >> 10;
    const int t  = bt & (SEQ - 1);
    float acc = cb[d];
    #pragma unroll
    for (int k = 0; k < DCONV; ++k) {
        const int back = (DCONV - 1) - k;
        if (t >= back)
            acc += xz[(size_t)(bt - back) * (2 * DINNER) + d] * cw[d * DCONV + k];
    }
    const float s = acc / (1.0f + __expf(-acc));
    xc [(size_t)bt * DINNER + d] = s;
    xch[(size_t)bt * DINNER + d] = (_Float16)s;
}

// ---------------------------------------------------------------------------
// Selective scan: one lane per (b,d) channel, 16 states in VGPRs, B/C tiles
// staged in LDS in 64-step chunks. Fuses +u*D and *silu(z); emits f16 for
// out_proj's WMMA A operand.
// ---------------------------------------------------------------------------
__global__ __launch_bounds__(256)
void scan_kernel(const float* __restrict__ u, const float* __restrict__ delta,
                 const float* __restrict__ xdbl, const float* __restrict__ xz,
                 const float* __restrict__ A_log, const float* __restrict__ Dp,
                 _Float16* __restrict__ y) {
    const int b = blockIdx.x >> 2;
    const int d = (blockIdx.x & 3) * 256 + threadIdx.x;

    float a[NSTATE], h[NSTATE];
    #pragma unroll
    for (int n = 0; n < NSTATE; ++n) {
        a[n] = -__expf(A_log[d * NSTATE + n]);
        h[n] = 0.0f;
    }
    const float Dv = Dp[d];

    __shared__ float BC[64][32];  // [t][0..15]=B, [16..31]=C
    for (int tc = 0; tc < SEQ / 64; ++tc) {
        #pragma unroll
        for (int j = 0; j < 8; ++j) {
            const int e  = threadIdx.x * 8 + j;
            const int tt = e >> 5;
            const int c  = e & 31;
            BC[tt][c] = xdbl[(size_t)((b << 10) + tc * 64 + tt) * 64 + DTRANK + c];
        }
        __syncthreads();
        for (int tt = 0; tt < 64; ++tt) {
            const int base = (b << 10) + tc * 64 + tt;
            const float dl = delta[(size_t)base * DINNER + d];
            const float uu = u[(size_t)base * DINNER + d];
            const float du = dl * uu;
            float accv = 0.0f;
            #pragma unroll
            for (int n = 0; n < NSTATE; ++n) {
                h[n] = __expf(dl * a[n]) * h[n] + du * BC[tt][n];
                accv += h[n] * BC[tt][NSTATE + n];
            }
            const float yv = accv + uu * Dv;
            const float z  = xz[(size_t)base * (2 * DINNER) + DINNER + d];
            y[(size_t)base * DINNER + d] = (_Float16)(yv * (z / (1.0f + __expf(-z))));
        }
        __syncthreads();
    }
}

// ---------------------------------------------------------------------------
extern "C" void kernel_launch(void* const* d_in, const int* in_sizes, int n_in,
                              void* d_out, int out_size, void* d_ws, size_t ws_size,
                              hipStream_t stream) {
    (void)in_sizes; (void)n_in; (void)out_size; (void)ws_size;

    const float* x         = (const float*)d_in[0];
    const float* embed_w   = (const float*)d_in[1];
    const float* embed_b   = (const float*)d_in[2];
    const float* norm_w    = (const float*)d_in[3];
    const float* in_proj_w = (const float*)d_in[4];
    const float* conv_w    = (const float*)d_in[5];
    const float* conv_b    = (const float*)d_in[6];
    const float* x_proj_w  = (const float*)d_in[7];
    const float* dt_proj_w = (const float*)d_in[8];
    const float* dt_proj_b = (const float*)d_in[9];
    const float* A_log     = (const float*)d_in[10];
    const float* D_param   = (const float*)d_in[11];
    const float* out_proj_w= (const float*)d_in[12];
    const float* norm_f_w  = (const float*)d_in[13];
    const float* dec_w     = (const float*)d_in[14];
    const float* dec_b     = (const float*)d_in[15];
    float* out = (float*)d_out;

    // ---- workspace carve-out (256B aligned blocks) ----
    char* p = (char*)d_ws;
    auto carve = [&](size_t bytes) {
        char* r = p;
        p += (bytes + 255) & ~(size_t)255;
        return r;
    };
    float*     hs     = (float*)    carve((size_t)NROWS * DMODEL * 4);
    float*     res    = (float*)    carve((size_t)NROWS * DMODEL * 4);
    _Float16*  hnh    = (_Float16*) carve((size_t)NROWS * DMODEL * 2);
    float*     xzb    = (float*)    carve((size_t)NROWS * 2 * DINNER * 4);
    float*     xconv  = (float*)    carve((size_t)NROWS * DINNER * 4);
    _Float16*  xconvh = (_Float16*) carve((size_t)NROWS * DINNER * 2);
    float*     xdbl   = (float*)    carve((size_t)NROWS * 64 * 4);
    _Float16*  xdblh  = (_Float16*) carve((size_t)NROWS * 64 * 2);
    float*     delta  = (float*)    carve((size_t)NROWS * DINNER * 4);
    _Float16*  ycoreh = (_Float16*) carve((size_t)NROWS * DINNER * 2);
    _Float16*  ipwh   = (_Float16*) carve((size_t)2 * 2 * DINNER * DMODEL * 2);
    _Float16*  xpwh   = (_Float16*) carve((size_t)2 * 64 * DINNER * 2);
    _Float16*  dtwh   = (_Float16*) carve((size_t)2 * DINNER * DTRANK * 2);
    _Float16*  opwh   = (_Float16*) carve((size_t)2 * DMODEL * DINNER * 2);
    _Float16*  dwh    = (_Float16*) carve((size_t)256 * DMODEL * 2);

    const dim3 blk(256);
    auto cvt = [&](const float* src, _Float16* dst, int n) {
        f32_to_f16_kernel<<<(n + 255) / 256, blk, 0, stream>>>(src, dst, n);
    };

    // 0) weights -> f16 (stays resident in L2)
    cvt(in_proj_w,  ipwh, 2 * 2 * DINNER * DMODEL);
    cvt(x_proj_w,   xpwh, 2 * 64 * DINNER);
    cvt(dt_proj_w,  dtwh, 2 * DINNER * DTRANK);
    cvt(out_proj_w, opwh, 2 * DMODEL * DINNER);
    cvt(dec_w,      dwh,  256 * DMODEL);

    // 1) embedding
    embed_kernel<<<NROWS, 512, 0, stream>>>(x, embed_w, embed_b, hs);

    for (int l = 0; l < 2; ++l) {
        const float*    nw  = norm_w    + (size_t)l * DMODEL;
        const _Float16* ipw = ipwh      + (size_t)l * 2 * DINNER * DMODEL;
        const float*    cw  = conv_w    + (size_t)l * DINNER * DCONV;
        const float*    cb  = conv_b    + (size_t)l * DINNER;
        const _Float16* xpw = xpwh      + (size_t)l * 64 * DINNER;
        const _Float16* dpw = dtwh      + (size_t)l * DINNER * DTRANK;
        const float*    dpb = dt_proj_b + (size_t)l * DINNER;
        const float*    al  = A_log     + (size_t)l * DINNER * NSTATE;
        const float*    dp  = D_param   + (size_t)l * DINNER;
        const _Float16* opw = opwh      + (size_t)l * DMODEL * DINNER;

        // 2) residual accumulate + RMSNorm (hn as f16)
        add_rmsnorm_kernel<<<NROWS, blk, 0, stream>>>(hs, res, hnh, nw, l == 0);
        // 3) in_proj: (4096,512) x (2048,512)^T -> xz (4096,2048)
        gemm_wmma<0><<<dim3(NROWS / 128, (2 * DINNER) / 64), blk, 0, stream>>>(
            hnh, DMODEL, ipw, DMODEL, xzb, 2 * DINNER, DMODEL, nullptr, nullptr);
        // 4) depthwise causal conv + SiLU
        conv_silu_kernel<<<(NROWS * DINNER) / 256, blk, 0, stream>>>(xzb, cw, cb, xconv, xconvh);
        // 5) x_proj: (4096,1024) x (64,1024)^T -> x_dbl (f32 for scan, f16 for dt GEMM)
        gemm_wmma<0><<<dim3(NROWS / 128, 64 / 64), blk, 0, stream>>>(
            xconvh, DINNER, xpw, DINNER, xdbl, 64, DINNER, nullptr, xdblh);
        // 6) dt_proj + softplus: (4096,32) x (1024,32)^T -> delta (4096,1024)
        gemm_wmma<1><<<dim3(NROWS / 128, DINNER / 64), blk, 0, stream>>>(
            xdblh, 64, dpw, DTRANK, delta, DINNER, DTRANK, dpb, nullptr);
        // 7) selective scan (+u*D, *silu(z)) -> ycore (f16)
        scan_kernel<<<BATCH * (DINNER / 256), blk, 0, stream>>>(
            xconv, delta, xdbl, xzb, al, dp, ycoreh);
        // 8) out_proj: (4096,1024) x (512,1024)^T -> hs (4096,512)
        gemm_wmma<0><<<dim3(NROWS / 128, DMODEL / 64), blk, 0, stream>>>(
            ycoreh, DINNER, opw, DINNER, hs, DMODEL, DINNER, nullptr, nullptr);
    }

    // 9) final residual + RMSNorm
    add_rmsnorm_kernel<<<NROWS, blk, 0, stream>>>(hs, res, hnh, norm_f_w, 0);
    // 10) decoder: (4096,512) x (256,512)^T + b -> out (4096,256)
    gemm_wmma<0><<<dim3(NROWS / 128, 256 / 64), blk, 0, stream>>>(
        hnh, DMODEL, dwh, DMODEL, out, 256, DMODEL, dec_b, nullptr);
}
